// DropLearner_71648644431894
// MI455X (gfx1250) — compile-verified, matrix operand order checked by reading, not run on hardware
//
#include <hip/hip_runtime.h>
#include <hip/hip_bf16.h>

typedef __attribute__((ext_vector_type(16))) _Float16 v16h;
typedef __attribute__((ext_vector_type(8)))  _Float16 v8h;
typedef __attribute__((ext_vector_type(8)))  float    v8f;

#define D_DIM   256   // embedding dim (K total)
#define H_DIM   64    // hidden dim
#define KSTEPS  8     // 256 / 32 (K per WMMA)
#define NTILES  4     // 64 / 16  (N tiles of hidden)
#define NFRAGS  (2 * NTILES * KSTEPS)        // 64 B-fragments (2 MLPs)
#define PACKED_HALVES (NFRAGS * 32 * 16)     // 32768 halves = 64 KB

// ---------------------------------------------------------------------------
// Pack w1_src / w1_dst (D x H, f32, row-major) into the CDNA5 16-bit
// B-matrix (32x16, K-major per lane) fragment layout, converted to f16.
//
// Chunked layout for conflict-free LDS reads: each fragment is two 16-byte
// chunks per lane; chunk c of fragment f for lane l lives at half-offset
//   ((f*2 + c)*32 + l) * 8
// so ds_load_b128 has a 16B lane stride -> lane i covers banks 4i..4i+3,
// conflict-free per 16-lane phase.
//
// B layout (16-bit, 32x16): lane%16 = N column; lanes 0-15 hold K = s*32 +
// {0..7, 16..23}; lanes 16-31 hold K = s*32 + {8..15, 24..31}.
// Fragment index f = (mlp*NTILES + t)*KSTEPS + s.
// ---------------------------------------------------------------------------
__global__ void pack_w1_kernel(const float* __restrict__ w1_src,
                               const float* __restrict__ w1_dst,
                               _Float16* __restrict__ packed) {
    int idx = blockIdx.x * blockDim.x + threadIdx.x;   // 0 .. NFRAGS*32-1
    if (idx >= NFRAGS * 32) return;
    int lane = idx & 31;
    int s    = (idx >> 5) & (KSTEPS - 1);
    int t    = (idx >> 8) & (NTILES - 1);
    int mlp  = idx >> 10;
    const float* w1 = mlp ? w1_dst : w1_src;
    int n     = t * 16 + (lane & 15);
    int kbase = s * 32 + ((lane < 16) ? 0 : 8);
    int frag  = idx >> 5;
    _Float16* dst = packed + ((size_t)frag * 64 + lane) * 8;   // chunk 0
    #pragma unroll
    for (int h = 0; h < 8; ++h) {
        dst[h]       = (_Float16)w1[(size_t)(kbase + h)      * H_DIM + n];
        dst[256 + h] = (_Float16)w1[(size_t)(kbase + 16 + h) * H_DIM + n]; // chunk 1
    }
}

__device__ inline void cvt8(v16h& a, int base, float4 x, float4 y) {
    a[base + 0] = (_Float16)x.x; a[base + 1] = (_Float16)x.y;
    a[base + 2] = (_Float16)x.z; a[base + 3] = (_Float16)x.w;
    a[base + 4] = (_Float16)y.x; a[base + 5] = (_Float16)y.y;
    a[base + 6] = (_Float16)y.z; a[base + 7] = (_Float16)y.w;
}

// ---------------------------------------------------------------------------
// Node MLP: one wave per 16-node tile; both MLPs share the A fragments.
// out[m] = relu(emb[m,:] @ w1 + b1) @ w2 + b2   for src and dst networks.
//
// - packed B staged once per block into LDS (shared by all 8 waves).
// - k-step outermost: only ONE A fragment live; 8 accumulators persist.
// - raw f32 of the next k-step's A is prefetched into fresh registers.
// - B fragments explicitly double-buffered in registers so each ds_load pair
//   is issued one WMMA ahead of its consumer (no s_wait_dscnt 0 stalls).
// ---------------------------------------------------------------------------
__global__ void __launch_bounds__(256)
node_mlp_kernel(const float* __restrict__ node_emb,
                const _Float16* __restrict__ packedB,
                const float* __restrict__ b1_src,
                const float* __restrict__ w2_src,
                const float* __restrict__ b2_src,
                const float* __restrict__ b1_dst,
                const float* __restrict__ w2_dst,
                const float* __restrict__ b2_dst,
                float* __restrict__ out_s,
                float* __restrict__ out_d,
                int n_nodes, int n_tiles) {
    __shared__ _Float16 ldsB[PACKED_HALVES];               // 64 KB
    const int tid = threadIdx.x;

    // ---- cooperative stage: 64 KB = 4096 uint4, 16 per thread ----
    {
        const uint4* g = (const uint4*)packedB;
        uint4*       l = (uint4*)ldsB;
        #pragma unroll
        for (int i = 0; i < 16; ++i)
            l[tid + 256 * i] = g[tid + 256 * i];
    }
    __syncthreads();

    const int lane = tid & 31;
    const int tile = blockIdx.x * (blockDim.x >> 5) + (tid >> 5);
    if (tile >= n_tiles) return;                 // wave-uniform exit, after barrier

    // ---- A fragment addressing (16-bit 16x32 layout) ----
    int m   = lane & 15;                         // A: M = lane % 16
    int row = tile * 16 + m;
    if (row >= n_nodes) row = n_nodes - 1;       // clamp (extra rows unused)
    const float* rowp = node_emb + (size_t)row * D_DIM;
    const int koff = (lane < 16) ? 0 : 8;

    const v8h* bchunks = (const v8h*)ldsB;

    v8f acc[8];                                  // j = mlp*4 + t
    #pragma unroll
    for (int j = 0; j < 8; ++j) acc[j] = (v8f){};

    // prologue: raw f32 for k-step 0
    float4 f0 = *(const float4*)(rowp + koff);
    float4 f1 = *(const float4*)(rowp + koff + 4);
    float4 f2 = *(const float4*)(rowp + koff + 16);
    float4 f3 = *(const float4*)(rowp + koff + 20);

    union BU { v16h v; v8h h[2]; };
    // prologue: B fragment (j=0, s=0)
    BU bcur;
    bcur.h[0] = bchunks[lane];
    bcur.h[1] = bchunks[lane + 32];

    #pragma unroll
    for (int s = 0; s < KSTEPS; ++s) {
        v16h a;
        cvt8(a, 0, f0, f1);
        cvt8(a, 8, f2, f3);
        if (s + 1 < KSTEPS) {                    // prefetch next k-step's A
            const float* nrp = rowp + (s + 1) * 32 + koff;
            f0 = *(const float4*)(nrp);
            f1 = *(const float4*)(nrp + 4);
            f2 = *(const float4*)(nrp + 16);
            f3 = *(const float4*)(nrp + 20);
        }
        #pragma unroll
        for (int j = 0; j < 8; ++j) {
            // issue the NEXT fragment's loads before consuming the current one
            BU bnext;
            if (!(s == KSTEPS - 1 && j == 7)) {
                int jn = (j + 1) & 7;
                int sn = (j == 7) ? s + 1 : s;
                int cn = ((jn * KSTEPS + sn) * 2) * 32 + lane;
                bnext.h[0] = bchunks[cn];
                bnext.h[1] = bchunks[cn + 32];
            } else {
                bnext = bcur;                    // dead value, keeps code simple
            }
            acc[j] = __builtin_amdgcn_wmma_f32_16x16x32_f16(
                        false, a, false, bcur.v, (short)0, acc[j], false, false);
            bcur = bnext;                        // pure rename after unroll
        }
    }

    // ---- epilogue: bias + relu + second layer + cross-lane reduce ----
    #pragma unroll
    for (int mlp = 0; mlp < 2; ++mlp) {
        const float* b1 = mlp ? b1_dst : b1_src;
        const float* w2 = mlp ? w2_dst : w2_src;
        const float  b2 = mlp ? b2_dst[0] : b2_src[0];
        float*      out = mlp ? out_d : out_s;

        float p[8];
        #pragma unroll
        for (int i = 0; i < 8; ++i) p[i] = 0.0f;

        #pragma unroll
        for (int t = 0; t < NTILES; ++t) {
            // C/D layout: VGPR i -> row (i | i+8), lane%16 -> column of tile t
            float b1v = b1[t * 16 + (lane & 15)];
            float w2v = w2[t * 16 + (lane & 15)];
            #pragma unroll
            for (int i = 0; i < 8; ++i) {
                float h = acc[mlp * NTILES + t][i] + b1v;
                h = h > 0.0f ? h : 0.0f;         // relu
                p[i] += h * w2v;
            }
        }

        // reduce across the 16 lanes of each half (columns of the tile row)
        #pragma unroll
        for (int i = 0; i < 8; ++i) {
            float v = p[i];
            #pragma unroll
            for (int mask = 1; mask < 16; mask <<= 1)
                v += __shfl_xor(v, mask, 32);
            int mm   = (lane < 16) ? i : (8 + i);
            int orow = tile * 16 + mm;
            if ((lane == 0 || lane == 16) && orow < n_nodes)
                out[orow] = v + b2;
        }
    }
}

// ---------------------------------------------------------------------------
// Edge gate: aug = sigmoid((logit(eps) + w_s[src] + w_d[dst]) / T), T = 0.5
// Per-block partial sums of aug for the deterministic mean reduction.
// ---------------------------------------------------------------------------
__global__ void edge_kernel(const int* __restrict__ src,
                            const int* __restrict__ dst,
                            const float* __restrict__ eps,
                            const float* __restrict__ ws,
                            const float* __restrict__ wd,
                            float* __restrict__ aug_out,
                            float* __restrict__ partials, int E) {
    __shared__ float red[256];
    int e = blockIdx.x * 256 + threadIdx.x;
    float aug = 0.0f;
    if (e < E) {
        float w  = ws[src[e]] + wd[dst[e]];
        float ep = eps[e];
        float g  = (logf(ep) - log1pf(-ep) + w) * 2.0f;   // / TEMPERATURE
        aug = 1.0f / (1.0f + expf(-g));
        aug_out[e] = aug;
    }
    red[threadIdx.x] = aug;
    __syncthreads();
    #pragma unroll
    for (int off = 128; off > 0; off >>= 1) {
        if (threadIdx.x < off) red[threadIdx.x] += red[threadIdx.x + off];
        __syncthreads();
    }
    if (threadIdx.x == 0) partials[blockIdx.x] = red[0];
}

// Deterministic final reduction: reg = 1 - sum(aug)/E
__global__ void reduce_kernel(const float* __restrict__ partials, int nparts,
                              float* __restrict__ reg_out, float inv_e) {
    __shared__ double red[256];
    double s = 0.0;
    for (int i = threadIdx.x; i < nparts; i += 256) s += (double)partials[i];
    red[threadIdx.x] = s;
    __syncthreads();
    #pragma unroll
    for (int off = 128; off > 0; off >>= 1) {
        if (threadIdx.x < off) red[threadIdx.x] += red[threadIdx.x + off];
        __syncthreads();
    }
    if (threadIdx.x == 0)
        *reg_out = 1.0f - (float)(red[0] * (double)inv_e);
}

// ---------------------------------------------------------------------------
extern "C" void kernel_launch(void* const* d_in, const int* in_sizes, int n_in,
                              void* d_out, int out_size, void* d_ws, size_t ws_size,
                              hipStream_t stream) {
    const float* node_emb = (const float*)d_in[0];
    const int*   src      = (const int*)d_in[1];
    const int*   dst      = (const int*)d_in[2];
    const float* eps      = (const float*)d_in[3];
    const float* w1_src   = (const float*)d_in[4];
    const float* b1_src   = (const float*)d_in[5];
    const float* w2_src   = (const float*)d_in[6];
    const float* b2_src   = (const float*)d_in[7];
    const float* w1_dst   = (const float*)d_in[8];
    const float* b1_dst   = (const float*)d_in[9];
    const float* w2_dst   = (const float*)d_in[10];
    const float* b2_dst   = (const float*)d_in[11];

    const int N = in_sizes[0] / D_DIM;
    const int E = in_sizes[1];
    float* out = (float*)d_out;          // out[0] = reg, out[1..E] = aug

    // workspace layout
    char*      ws       = (char*)d_ws;
    _Float16*  packedB  = (_Float16*)ws;                    // 64 KB
    float*     wnode_s  = (float*)(ws + PACKED_HALVES * sizeof(_Float16));
    float*     wnode_d  = wnode_s + N;
    float*     partials = wnode_d + N;

    pack_w1_kernel<<<(NFRAGS * 32 + 255) / 256, 256, 0, stream>>>(w1_src, w1_dst, packedB);

    const int n_tiles = (N + 15) / 16;                      // 16-node tiles
    const int wavesPerBlk = 8;                              // 256 threads
    node_mlp_kernel<<<(n_tiles + wavesPerBlk - 1) / wavesPerBlk, 32 * wavesPerBlk, 0, stream>>>(
        node_emb, packedB,
        b1_src, w2_src, b2_src,
        b1_dst, w2_dst, b2_dst,
        wnode_s, wnode_d, N, n_tiles);

    const int eblocks = (E + 255) / 256;
    edge_kernel<<<eblocks, 256, 0, stream>>>(src, dst, eps, wnode_s, wnode_d,
                                             out + 1, partials, E);
    reduce_kernel<<<1, 256, 0, stream>>>(partials, eblocks, out, 1.0f / (float)E);
}
